// TimeLSTM_55396488184541
// MI455X (gfx1250) — matrix operand order repaired and checked by reading, not compiled, not run on hardware
//
#include <hip/hip_runtime.h>
#include <hip/hip_bf16.h>
#include <math.h>

#define NU 2048
#define DH 512
#define NG 2560   // 5 gates * 512
#define TS 64

typedef __attribute__((ext_vector_type(16))) __bf16        v16bf;
typedef __attribute__((ext_vector_type(8)))  float         v8f;
typedef __attribute__((ext_vector_type(8)))  unsigned int  v8u;

// ---------- helpers ----------
__device__ __forceinline__ unsigned pkbf(float lo, float hi) {
  unsigned ul = __builtin_bit_cast(unsigned, lo);
  unsigned uh = __builtin_bit_cast(unsigned, hi);
  ul = (ul + 0x7FFFu + ((ul >> 16) & 1u)) >> 16;      // RNE bf16
  uh = (uh + 0x7FFFu + ((uh >> 16) & 1u)) & 0xFFFF0000u;
  return (ul & 0xFFFFu) | uh;
}
__device__ __forceinline__ unsigned short bf16s(float x) {
  unsigned u = __builtin_bit_cast(unsigned, x);
  u = (u + 0x7FFFu + ((u >> 16) & 1u)) >> 16;
  return (unsigned short)u;
}
__device__ __forceinline__ float sigf(float x) { return 1.0f / (1.0f + __expf(-x)); }

// element (row, j) of a [rows x 512] matrix -> bf16 A-fragment dword index
// frag(kc, mtile): 32 lanes x 8 dwords; lane = row%16 + 16*((kj>>3)&1);
// dword = ((kj&7)>>1) + (kj&16 ? 4 : 0); half = kj&1   (kj = j%32, kc = j/32)
__device__ __forceinline__ size_t afrag_dw(int row, int j) {
  int kc = j >> 5, kj = j & 31;
  int lane = (row & 15) + (((kj >> 3) & 1) << 4);
  int d = ((kj & 7) >> 1) + ((kj & 16) ? 4 : 0);
  return (((size_t)kc * 128 + (row >> 4)) * 32 + lane) * 8 + d;
}

// ---------- weight / misc packing ----------
__device__ __forceinline__ float wbig_at(int k, int g, int j,
    const float* Wxi, const float* Wxf, const float* Wxc, const float* Wxo, const float* Wxt,
    const float* Whi, const float* Whf, const float* Whc, const float* Who) {
  if (k < 512) {
    const float* W = (g == 0) ? Wxi : (g == 1) ? Wxf : (g == 2) ? Wxc : (g == 3) ? Wxo : Wxt;
    return W[(size_t)k * DH + j];
  } else {
    if (g == 4) return 0.0f;  // T gate has no h term
    const float* W = (g == 0) ? Whi : (g == 1) ? Whf : (g == 2) ? Whc : Who;
    return W[(size_t)(k - 512) * DH + j];
  }
}

__global__ __launch_bounds__(256) void pack_wbig(
    const float* Wxi, const float* Wxf, const float* Wxc, const float* Wxo, const float* Wxt,
    const float* Whi, const float* Whf, const float* Whc, const float* Who,
    unsigned* __restrict__ WF) {
  int tid = blockIdx.x * 256 + threadIdx.x;       // 32 KC * 160 NT * 32 lanes = 163840
  int kc = tid / 5120;
  int rem = tid % 5120;
  int nt = rem >> 5;
  int lane = tid & 31;
  int n = nt * 16 + (lane & 15);
  int g = n >> 9, j = n & 511;
  int kb = kc * 32 + ((lane & 16) ? 8 : 0);
  unsigned* out = WF + ((size_t)(kc * 160 + nt) * 32 + lane) * 8;
#pragma unroll
  for (int d = 0; d < 8; ++d) {
    float v0 = wbig_at(kb + 2 * (d & 3) + ((d & 4) << 2),     g, j, Wxi, Wxf, Wxc, Wxo, Wxt, Whi, Whf, Whc, Who);
    float v1 = wbig_at(kb + 2 * (d & 3) + ((d & 4) << 2) + 1, g, j, Wxi, Wxf, Wxc, Wxo, Wxt, Whi, Whf, Whc, Who);
    out[d] = pkbf(v0, v1);
  }
}

__global__ __launch_bounds__(256) void pack_wco(const float* __restrict__ W,
                                                unsigned* __restrict__ WF) {
  int tid = blockIdx.x * 256 + threadIdx.x;       // 16 KC * 32 NT * 32 lanes = 16384
  int lane = tid & 31;
  int nt = (tid >> 5) & 31;
  int kc = tid >> 10;
  int n = nt * 16 + (lane & 15);
  int kb = kc * 32 + ((lane & 16) ? 8 : 0);
  unsigned* out = WF + ((size_t)(kc * 32 + nt) * 32 + lane) * 8;
#pragma unroll
  for (int d = 0; d < 8; ++d)
    out[d] = pkbf(W[(size_t)(kb + 2 * (d & 3) + ((d & 4) << 2)) * DH + n], W[(size_t)(kb + 2 * (d & 3) + ((d & 4) << 2) + 1) * DH + n]);
}

__global__ __launch_bounds__(256) void pack_misc(
    const float* bxi, const float* bhi, const float* bxf, const float* bhf,
    const float* bxt, const float* bxc, const float* bhc, const float* bxo,
    const float* bho, const float* bco,
    const float* wtt, const float* btt, const float* wto, const float* bto,
    const float* dt, float* __restrict__ bias, float* __restrict__ Ttt,
    float* __restrict__ Tto) {
  int tid = blockIdx.x * 256 + threadIdx.x;       // 32768 threads
  if (tid < NG) {
    int g = tid >> 9, j = tid & 511;
    float v;
    if (g == 0) v = bxi[j] + bhi[j];
    else if (g == 1) v = bxf[j] + bhf[j];
    else if (g == 2) v = bxc[j] + bhc[j];
    else if (g == 3) v = bxo[j] + bho[j] + bco[j];  // fold Wco bias into o-gate
    else v = bxt[j];
    bias[tid] = v;
  }
  int t = tid >> 9, j = tid & 511;
  float d = dt[t];
  Ttt[tid] = sigf(d * wtt[j] + btt[j]);
  Tto[tid] = d * wto[j] + bto[j];
}

// ---------- pack x_t into A-fragments ----------
__global__ __launch_bounds__(256) void pack_x(const float* __restrict__ X,
                                              unsigned* __restrict__ XF) {
  int tid = blockIdx.x * 256 + threadIdx.x;       // NU*512/2 = 524288
  int row = tid >> 8;
  int j0 = (tid & 255) * 2;
  float2 v = *(const float2*)(X + (size_t)row * DH + j0);
  XF[afrag_dw(row, j0)] = pkbf(v.x, v.y);
}

// ---------- S1: [x_t ; h] (2048x1024) @ Wbig (1024x2560), fused gate activations ----------
__global__ __launch_bounds__(256) void s1_gemm(
    const unsigned* __restrict__ XF, const unsigned* __restrict__ HF,
    const unsigned* __restrict__ WF, const float* __restrict__ bias,
    const float* __restrict__ ttt, const float* __restrict__ tto,
    float* __restrict__ G, int kcmax) {
  const int lane = threadIdx.x & 31;
  const int wv = threadIdx.x >> 5;
  const int gw = blockIdx.x * 8 + wv;             // 2560 waves = 64 M x 40 N
  const int mi = gw & 63;
  const int ni = gw >> 6;
  const int rm = mi * 32;
  const int cn = ni * 64;
  v8f acc[2][4] = {};
  const v8u* bfp = (const v8u*)WF;
  for (int kc = 0; kc < kcmax; ++kc) {
    const v8u* afp = (const v8u*)((kc < 16) ? XF : HF);
    int kcl = (kc < 16) ? kc : (kc - 16);
    v16bf a0 = __builtin_bit_cast(v16bf, afp[((size_t)kcl * 128 + mi * 2) * 32 + lane]);
    v16bf a1 = __builtin_bit_cast(v16bf, afp[((size_t)kcl * 128 + mi * 2 + 1) * 32 + lane]);
#pragma unroll
    for (int nt = 0; nt < 4; ++nt) {
      v16bf b = __builtin_bit_cast(v16bf, bfp[((size_t)kc * 160 + (cn >> 4) + nt) * 32 + lane]);
      acc[0][nt] = __builtin_amdgcn_wmma_f32_16x16x32_bf16(false, a0, false, b, (short)0,
                                                           acc[0][nt], false, false);
      acc[1][nt] = __builtin_amdgcn_wmma_f32_16x16x32_bf16(false, a1, false, b, (short)0,
                                                           acc[1][nt], false, false);
    }
  }
  const int lrow = (lane >> 4) * 8;
  const int lcol = lane & 15;
#pragma unroll
  for (int nt = 0; nt < 4; ++nt) {
    int col = cn + nt * 16 + lcol;
    float add = bias[col];
    int act = 0;                                  // 0=sigmoid, 1=tanh, 2=linear
    if (col < 1024) act = 0;                      // i, f
    else if (col < 1536) act = 1;                 // c_hat
    else if (col < 2048) { act = 2; add += tto[col - 1536]; }  // o_pre (linear)
    else { act = 0; add += ttt[col - 2048]; }     // T gate
#pragma unroll
    for (int mt = 0; mt < 2; ++mt) {
#pragma unroll
      for (int r = 0; r < 8; ++r) {
        float v = acc[mt][nt][r] + add;
        v = (act == 0) ? sigf(v) : (act == 1) ? tanhf(v) : v;
        int row = rm + mt * 16 + lrow + r;
        G[(size_t)row * NG + col] = v;
      }
    }
  }
}

// ---------- E1: cell update + c-fragment pack, fused with packing x_{t+1} ----------
__global__ __launch_bounds__(256) void e1_cell(
    const float* __restrict__ G, float* __restrict__ Cbuf, unsigned* __restrict__ CF,
    const float* __restrict__ Xnext, unsigned* __restrict__ XF, int first) {
  int tid = blockIdx.x * 256 + threadIdx.x;       // 2 * 524288
  int half = tid >> 19;
  int id = tid & 524287;
  int row = id >> 8;
  int j0 = (id & 255) * 2;
  if (half == 0) {
    const float* g = G + (size_t)row * NG + j0;
    float i0 = g[0],    i1 = g[1];
    float f0 = g[512],  f1 = g[513];
    float h0 = g[1024], h1 = g[1025];
    float T0 = g[2048], T1 = g[2049];
    size_t ci = (size_t)row * DH + j0;
    float cp0 = first ? 0.0f : Cbuf[ci];
    float cp1 = first ? 0.0f : Cbuf[ci + 1];
    float c0 = f0 * cp0 + i0 * T0 * h0;
    float c1 = f1 * cp1 + i1 * T1 * h1;
    Cbuf[ci] = c0;
    Cbuf[ci + 1] = c1;
    CF[afrag_dw(row, j0)] = pkbf(c0, c1);
  } else if (Xnext) {
    float2 v = *(const float2*)(Xnext + (size_t)row * DH + j0);
    XF[afrag_dw(row, j0)] = pkbf(v.x, v.y);
  }
}

// ---------- S2: c_t (2048x512) @ Wco (512x512), fused o-gate + h, pack h-fragments ----------
__global__ __launch_bounds__(256) void s2_gemm(
    const unsigned* __restrict__ CF, const unsigned* __restrict__ WF,
    const float* __restrict__ G, const float* __restrict__ Cbuf,
    float* __restrict__ Hout, unsigned short* __restrict__ HFout) {
  const int lane = threadIdx.x & 31;
  const int wv = threadIdx.x >> 5;
  const int gw = blockIdx.x * 8 + wv;             // 512 waves = 64 M x 8 N
  const int mi = gw & 63;
  const int ni = gw >> 6;
  const int rm = mi * 32;
  const int cn = ni * 64;
  v8f acc[2][4] = {};
  const v8u* afp = (const v8u*)CF;
  const v8u* bfp = (const v8u*)WF;
  for (int kc = 0; kc < 16; ++kc) {
    v16bf a0 = __builtin_bit_cast(v16bf, afp[((size_t)kc * 128 + mi * 2) * 32 + lane]);
    v16bf a1 = __builtin_bit_cast(v16bf, afp[((size_t)kc * 128 + mi * 2 + 1) * 32 + lane]);
#pragma unroll
    for (int nt = 0; nt < 4; ++nt) {
      v16bf b = __builtin_bit_cast(v16bf, bfp[((size_t)kc * 32 + (cn >> 4) + nt) * 32 + lane]);
      acc[0][nt] = __builtin_amdgcn_wmma_f32_16x16x32_bf16(false, a0, false, b, (short)0,
                                                           acc[0][nt], false, false);
      acc[1][nt] = __builtin_amdgcn_wmma_f32_16x16x32_bf16(false, a1, false, b, (short)0,
                                                           acc[1][nt], false, false);
    }
  }
  const int lrow = (lane >> 4) * 8;
  const int lcol = lane & 15;
#pragma unroll
  for (int nt = 0; nt < 4; ++nt) {
    int col = cn + nt * 16 + lcol;
#pragma unroll
    for (int mt = 0; mt < 2; ++mt) {
#pragma unroll
      for (int r = 0; r < 8; ++r) {
        int row = rm + mt * 16 + lrow + r;
        float opre = G[(size_t)row * NG + 1536 + col];
        float o = sigf(acc[mt][nt][r] + opre);
        float cv = Cbuf[(size_t)row * DH + col];
        float h = o * tanhf(cv);
        Hout[(size_t)row * DH + col] = h;
        size_t di = afrag_dw(row, col);
        HFout[di * 2 + (col & 1)] = bf16s(h);
      }
    }
  }
}

// ---------- final: time-weighted aggregation with softmax(-cumsum(dt)) ----------
__global__ __launch_bounds__(256) void agg_kernel(const float* __restrict__ dt,
                                                  const float* __restrict__ HS,
                                                  float* __restrict__ out) {
  int idx = blockIdx.x * 256 + threadIdx.x;       // NU*DH
  float cum = 0.f, mx = -1e30f;
  for (int t = 0; t < TS; ++t) { cum += dt[t]; mx = fmaxf(mx, -cum); }
  float s = 0.f; cum = 0.f;
  for (int t = 0; t < TS; ++t) { cum += dt[t]; s += __expf(-cum - mx); }
  float acc = 0.f; cum = 0.f;
  for (int t = 0; t < TS; ++t) {
    cum += dt[t];
    acc += __expf(-cum - mx) * HS[(size_t)t * (NU * DH) + idx];
  }
  out[idx] = acc / s;
}

// ---------- host launch ----------
extern "C" void kernel_launch(void* const* d_in, const int* in_sizes, int n_in,
                              void* d_out, int out_size, void* d_ws, size_t ws_size,
                              hipStream_t stream) {
  const float* emb = (const float*)d_in[0];
  const float* dt  = (const float*)d_in[1];
  const float* Wxi = (const float*)d_in[2];  const float* bxi = (const float*)d_in[3];
  const float* Whi = (const float*)d_in[4];  const float* bhi = (const float*)d_in[5];
  const float* Wxf = (const float*)d_in[6];  const float* bxf = (const float*)d_in[7];
  const float* Whf = (const float*)d_in[8];  const float* bhf = (const float*)d_in[9];
  const float* Wxt = (const float*)d_in[10]; const float* bxt = (const float*)d_in[11];
  const float* Wtt = (const float*)d_in[12]; const float* btt = (const float*)d_in[13];
  const float* Wxc = (const float*)d_in[14]; const float* bxc = (const float*)d_in[15];
  const float* Whc = (const float*)d_in[16]; const float* bhc = (const float*)d_in[17];
  const float* Wxo = (const float*)d_in[18]; const float* bxo = (const float*)d_in[19];
  const float* Wto = (const float*)d_in[20]; const float* bto = (const float*)d_in[21];
  const float* Who = (const float*)d_in[22]; const float* bho = (const float*)d_in[23];
  const float* Wco = (const float*)d_in[24]; const float* bco = (const float*)d_in[25];

  char* ws = (char*)d_ws;
  unsigned* WBF  = (unsigned*)(ws + 0);           //  5,242,880 B  bf16 Wbig frags
  unsigned* WCF  = (unsigned*)(ws + 5242880);     //    524,288 B  bf16 Wco frags
  float*    BIAS = (float*)   (ws + 5767168);     //     10,240 B
  float*    TTT  = (float*)   (ws + 5777408);     //    131,072 B
  float*    TTO  = (float*)   (ws + 5908480);     //    131,072 B
  float*    Gbuf = (float*)   (ws + 6039552);     // 20,971,520 B  gate pre/post activations
  float*    Cbuf = (float*)   (ws + 27011072);    //  4,194,304 B  c_t fp32
  unsigned* CF   = (unsigned*)(ws + 31205376);    //  2,097,152 B  c_t bf16 frags
  unsigned* XF   = (unsigned*)(ws + 33302528);    //  2,097,152 B  x_t bf16 frags
  unsigned* HF   = (unsigned*)(ws + 35399680);    //  2,097,152 B  h_t bf16 frags

  float* out_agg    = (float*)d_out;
  float* out_hidden = out_agg + (size_t)NU * DH;

  pack_wbig<<<640, 256, 0, stream>>>(Wxi, Wxf, Wxc, Wxo, Wxt, Whi, Whf, Whc, Who, WBF);
  pack_wco<<<64, 256, 0, stream>>>(Wco, WCF);
  pack_misc<<<128, 256, 0, stream>>>(bxi, bhi, bxf, bhf, bxt, bxc, bhc, bxo, bho, bco,
                                     Wtt, btt, Wto, bto, dt, BIAS, TTT, TTO);
  pack_x<<<2048, 256, 0, stream>>>(emb, XF);

  for (int t = 0; t < TS; ++t) {
    const unsigned* hf = (t == 0) ? nullptr : HF;
    int kcmax = (t == 0) ? 16 : 32;
    s1_gemm<<<320, 256, 0, stream>>>(XF, hf, WBF, BIAS, TTT + t * DH, TTO + t * DH,
                                     Gbuf, kcmax);
    const float* xnext = (t + 1 < TS) ? (emb + (size_t)(t + 1) * NU * DH) : nullptr;
    e1_cell<<<4096, 256, 0, stream>>>(Gbuf, Cbuf, CF, xnext, XF, t == 0 ? 1 : 0);
    s2_gemm<<<64, 256, 0, stream>>>(CF, WCF, Gbuf, Cbuf,
                                    out_hidden + (size_t)t * NU * DH,
                                    (unsigned short*)HF);
  }
  agg_kernel<<<4096, 256, 0, stream>>>(dt, out_hidden, out_agg);
}
